// HFE_74302934221277
// MI455X (gfx1250) — matrix operand order repaired and checked by reading, not compiled, not run on hardware
//
#include <hip/hip_runtime.h>
#include <cstdint>
#include <cstddef>

// ---------------------------------------------------------------------------
// CDNA5 (gfx1250) implicit-GEMM deformable-conv pipeline, bf16 WMMA (wave32)
// Tap-major K ordering + [row][x][cin] LDS slab => A fragment = 2x ds_load_b128
// M=32 per wave (2 A fragments) => B-fragment loads amortized over 2 WMMAs
// ---------------------------------------------------------------------------

typedef __attribute__((ext_vector_type(16))) __bf16 v16bf;
typedef __attribute__((ext_vector_type(8)))  float  v8f;

union FragBF {
  v16bf bf;
  unsigned int u[8];
  uint4 q[2];
};

__device__ __forceinline__ unsigned short f2bf(float f) {
  unsigned int u = __float_as_uint(f);
  u += 0x7FFFu + ((u >> 16) & 1u);          // round-to-nearest-even
  return (unsigned short)(u >> 16);
}

template <bool RB>
__device__ __forceinline__ v8f wmma_bf16(const FragBF& a, const FragBF& b, v8f c) {
  // D = A(16x32 bf16) * B(32x16 bf16) + C(16x16 f32)
  return __builtin_amdgcn_wmma_f32_16x16x32_bf16(
      /*neg_a=*/false, a.bf, /*neg_b=*/false, b.bf,
      /*c_mod=*/(short)0, c, /*reuse_a=*/false, /*reuse_b=*/RB);
}

// A-fragment K-offset (vgpr v, lane half) per 16-bit A 16x32 ISA layout
__device__ __forceinline__ int a_koff(int v, int half) {
  return (v < 4) ? (half * 8 + 2 * v) : (16 + half * 8 + 2 * (v - 4));
}

// ---------------------------------------------------------------------------
// Weight prepack: f32 -> bf16 in exact B-fragment layout, tap-major K.
//   dst[((nt*KT + kt)*32 + lane)*16 + e] = W[K = kt*32 + (lane>>4)*8 + (e&7) + (e>>3)*16,
//                                            N = nt*16 + (lane&15)]
//   K = tap*Cin + cin
// mode 2: W = w[o][cin][ky][kx]              (conv / offset conv / deform)
// mode 3: W = w[cin][o][2-ky][2-kx]          (deconv: flip + transpose)
// ---------------------------------------------------------------------------
__global__ void pack_weights(const float* __restrict__ w, unsigned short* __restrict__ dst,
                             int Cin, int Cout, int mode, int total) {
  int idx = blockIdx.x * 256 + threadIdx.x;
  if (idx >= total) return;
  const int KT = Cin * 9 / 32;
  int e    = idx & 15;
  int lane = (idx >> 4) & 31;
  int r    = idx >> 9;
  int kt   = r % KT;
  int nt   = r / KT;
  int o    = nt * 16 + (lane & 15);
  int K    = kt * 32 + (lane >> 4) * 8 + (e & 7) + (e >> 3) * 16;
  float val = 0.f;
  if (o < Cout) {
    int tap = K / Cin, cin = K % Cin;
    size_t si;
    if (mode == 3) {
      int ky = tap / 3, kx = tap % 3;
      si = (((size_t)cin * Cout + o) * 3 + (2 - ky)) * 3 + (2 - kx);
    } else {
      si = ((size_t)o * Cin + cin) * 9 + tap;
    }
    val = w[si];
  }
  dst[idx] = f2bf(val);
}

// ---------------------------------------------------------------------------
// 3x3 conv (pad=1), implicit GEMM with bf16 WMMA. One wave per 32-wide x tile.
// LDS slab layout [row(3)][x(34)][cin] bf16 -> A fragment = 2x ds_load_b128.
// Cin=128 is a virtual concat: channels [0,64) from srcA, [64,128) from srcB.
// ---------------------------------------------------------------------------
template <int CIN, int NTILES, bool RELU>
__global__ __launch_bounds__(128) void conv3_wmma(
    const float* __restrict__ srcA, const float* __restrict__ srcB,
    const unsigned short* __restrict__ wpack, const float* __restrict__ bias,
    float* __restrict__ out, int Cout) {
  constexpr int KT  = CIN * 9 / 32;
  constexpr int CPT = CIN / 32;            // K-chunks per tap
  constexpr int MW  = 34;                  // 32 + 2 halo
  const int lane = threadIdx.x & 31;
  const int wid  = threadIdx.x >> 5;
  const int tile = blockIdx.x * 4 + wid;   // B*H*(W/32) = 4096 tiles
  const int x0 = (tile & 3) * 32;
  const int y  = (tile >> 2) & 127;
  const int b  = tile >> 9;

  extern __shared__ __align__(16) unsigned short smem[];
  unsigned short* s = smem + (size_t)wid * 3 * MW * CIN;

  // Stage halo slab transposed: s[(row*MW + xx)*CIN + c]; lane <-> cin so the
  // per-(row,xx) channel run is LDS-contiguous (global lines L0-cached).
  for (int rxx = 0; rxx < 3 * MW; ++rxx) {
    int row = rxx / MW, xx = rxx % MW;
    int gy = y + row - 1, gx = x0 + xx - 1;
    bool inb = ((unsigned)gy < 128u) && ((unsigned)gx < 128u);
    for (int c = lane; c < CIN; c += 32) {
      float v = 0.f;
      if (inb) {
        const float* src = (c < 64) ? srcA : srcB;
        v = src[(((size_t)b * 64 + (c & 63)) * 128 + gy) * 128 + gx];
      }
      s[rxx * CIN + c] = f2bf(v);
    }
  }
  __syncthreads();

  v8f acc[2][NTILES];
#pragma unroll
  for (int j = 0; j < 2; ++j)
#pragma unroll
    for (int nt = 0; nt < NTILES; ++nt)
#pragma unroll
      for (int i = 0; i < 8; ++i) acc[j][nt][i] = 0.f;

  const int half = lane >> 4;
  const int ml   = lane & 15;

  for (int kt = 0; kt < KT; ++kt) {
    const int tap = kt / CPT;
    const int cb  = (kt % CPT) * 32;
    const int ky = tap / 3, kx = tap % 3;

    FragBF a[2];
#pragma unroll
    for (int j = 0; j < 2; ++j) {
      int m = j * 16 + ml;
      const unsigned short* p = s + ((size_t)(ky * MW + m + kx)) * CIN + cb + half * 8;
      a[j].q[0] = *(const uint4*)p;          // cin = cb+half*8 .. +7   (K 0..7 | 8..15)
      a[j].q[1] = *(const uint4*)(p + 16);   // cin = +16 run           (K 16..23 | 24..31)
    }
#pragma unroll
    for (int nt = 0; nt < NTILES; ++nt) {
      FragBF bf;
      const uint4* wp = (const uint4*)(wpack + ((size_t)(nt * KT + kt) * 32 + lane) * 16);
      bf.q[0] = wp[0];
      bf.q[1] = wp[1];
      acc[0][nt] = wmma_bf16<false>(a[0], bf, acc[0][nt]);
      acc[1][nt] = wmma_bf16<true >(a[1], bf, acc[1][nt]);   // same B: reuse hint
    }
  }

  // Epilogue: D element (M = r + 8*half, N = lane&15)
#pragma unroll
  for (int j = 0; j < 2; ++j)
#pragma unroll
    for (int nt = 0; nt < NTILES; ++nt) {
      int o = nt * 16 + ml;
      if (o < Cout) {
        float bv = bias[o];
        float* op = out + (((size_t)b * Cout + o) * 128 + y) * 128 + x0 + j * 16 + half * 8;
#pragma unroll
        for (int r = 0; r < 8; ++r) {
          float v = acc[j][nt][r] + bv;
          if (RELU) v = fmaxf(v, 0.f);
          op[r] = v;
        }
      }
    }
}

// ---------------------------------------------------------------------------
// Deformable 3x3 conv: bilinear-gather A fragments (tap-major K), bf16 WMMA.
// Offsets staged in LDS; per-channel gathers hit L2 (inputs << 192MB L2).
// ---------------------------------------------------------------------------
template <int CIN, bool RELU>
__global__ __launch_bounds__(128) void dconv_wmma(
    const float* __restrict__ srcA, const float* __restrict__ srcB,
    const float* __restrict__ offs, const unsigned short* __restrict__ wpack,
    const float* __restrict__ bias, float* __restrict__ out, int Cout) {
  constexpr int KT  = CIN * 9 / 32;
  constexpr int CPT = CIN / 32;
  const int lane = threadIdx.x & 31;
  const int wid  = threadIdx.x >> 5;
  const int tile = blockIdx.x * 4 + wid;
  const int x0 = (tile & 3) * 32;
  const int y  = (tile >> 2) & 127;
  const int b  = tile >> 9;

  __shared__ float soff[4][18][32];
  for (int i = lane; i < 18 * 32; i += 32) {
    int ch = i / 32, xx = i % 32;
    soff[wid][ch][xx] = offs[(((size_t)b * 18 + ch) * 128 + y) * 128 + x0 + xx];
  }
  __syncthreads();

  v8f acc[2][4];
#pragma unroll
  for (int j = 0; j < 2; ++j)
#pragma unroll
    for (int nt = 0; nt < 4; ++nt)
#pragma unroll
      for (int i = 0; i < 8; ++i) acc[j][nt][i] = 0.f;

  const int half = lane >> 4;
  const int ml   = lane & 15;

  for (int kt = 0; kt < KT; ++kt) {
    const int tap = kt / CPT;
    const int cb  = (kt % CPT) * 32;
    const int ky = tap / 3, kx = tap % 3;

    FragBF a[2];
#pragma unroll
    for (int j = 0; j < 2; ++j) {
      const int m = j * 16 + ml;
      float py = (float)(y + ky - 1) + soff[wid][tap * 2 + 0][m];
      float px = (float)(x0 + m + kx - 1) + soff[wid][tap * 2 + 1][m];
      float yf = floorf(py), xf = floorf(px);
      float fy = py - yf, fx = px - xf;
      int yi0 = (int)yf, xi0 = (int)xf;
      int yi1 = yi0 + 1, xi1 = xi0 + 1;
      float v00 = ((unsigned)yi0 < 128u && (unsigned)xi0 < 128u) ? 1.f : 0.f;
      float v01 = ((unsigned)yi0 < 128u && (unsigned)xi1 < 128u) ? 1.f : 0.f;
      float v10 = ((unsigned)yi1 < 128u && (unsigned)xi0 < 128u) ? 1.f : 0.f;
      float v11 = ((unsigned)yi1 < 128u && (unsigned)xi1 < 128u) ? 1.f : 0.f;
      float w00 = (1.f - fy) * (1.f - fx) * v00;
      float w01 = (1.f - fy) * fx * v01;
      float w10 = fy * (1.f - fx) * v10;
      float w11 = fy * fx * v11;
      int yc0 = min(max(yi0, 0), 127), yc1 = min(max(yi1, 0), 127);
      int xc0 = min(max(xi0, 0), 127), xc1 = min(max(xi1, 0), 127);
      int i00 = yc0 * 128 + xc0, i01 = yc0 * 128 + xc1;
      int i10 = yc1 * 128 + xc0, i11 = yc1 * 128 + xc1;

#pragma unroll
      for (int v = 0; v < 8; ++v) {
        int c0 = cb + a_koff(v, half);
        int c1 = c0 + 1;
        const float* p0 = ((c0 < 64) ? srcA : srcB) + ((size_t)b * 64 + (c0 & 63)) * 16384;
        const float* p1 = ((c1 < 64) ? srcA : srcB) + ((size_t)b * 64 + (c1 & 63)) * 16384;
        float s0 = w00 * p0[i00] + w01 * p0[i01] + w10 * p0[i10] + w11 * p0[i11];
        float s1 = w00 * p1[i00] + w01 * p1[i01] + w10 * p1[i10] + w11 * p1[i11];
        a[j].u[v] = (unsigned int)f2bf(s0) | ((unsigned int)f2bf(s1) << 16);
      }
    }
#pragma unroll
    for (int nt = 0; nt < 4; ++nt) {
      FragBF bf;
      const uint4* wp = (const uint4*)(wpack + ((size_t)(nt * KT + kt) * 32 + lane) * 16);
      bf.q[0] = wp[0];
      bf.q[1] = wp[1];
      acc[0][nt] = wmma_bf16<false>(a[0], bf, acc[0][nt]);
      acc[1][nt] = wmma_bf16<true >(a[1], bf, acc[1][nt]);
    }
  }

#pragma unroll
  for (int j = 0; j < 2; ++j)
#pragma unroll
    for (int nt = 0; nt < 4; ++nt) {
      int o = nt * 16 + ml;
      if (o < Cout) {
        float bv = bias[o];
        float* op = out + (((size_t)b * Cout + o) * 128 + y) * 128 + x0 + j * 16 + half * 8;
#pragma unroll
        for (int r = 0; r < 8; ++r) {
          float v = acc[j][nt][r] + bv;
          if (RELU) v = fmaxf(v, 0.f);
          op[r] = v;
        }
      }
    }
}

__global__ void addk(const float* __restrict__ a, const float* __restrict__ b,
                     float* __restrict__ o, int n) {
  int i = blockIdx.x * 256 + threadIdx.x;
  if (i < n) o[i] = a[i] + b[i];
}

// ---------------------------------------------------------------------------
extern "C" void kernel_launch(void* const* d_in, const int* in_sizes, int n_in,
                              void* d_out, int out_size, void* d_ws, size_t ws_size,
                              hipStream_t stream) {
  (void)in_sizes; (void)n_in; (void)out_size; (void)ws_size;
  const float* x_in      = (const float*)d_in[0];
  const float* deconv_w  = (const float*)d_in[1];
  const float* deconv_b  = (const float*)d_in[2];
  const float* off_w     = (const float*)d_in[3];
  const float* off_b     = (const float*)d_in[4];
  const float* dcn_w     = (const float*)d_in[5];
  const float* dcn_b     = (const float*)d_in[6];
  const float* fin_off_w = (const float*)d_in[7];
  const float* fin_off_b = (const float*)d_in[8];
  const float* fin_dcn_w = (const float*)d_in[9];
  const float* fin_dcn_b = (const float*)d_in[10];

  const size_t BIG  = (size_t)8 * 64 * 128 * 128 * sizeof(float);   // 33.5 MB
  const size_t OFFB = (size_t)8 * 18 * 128 * 128 * sizeof(float);   //  9.4 MB
  char* ws = (char*)d_ws;
  float* B0  = (float*)(ws);
  float* B1  = (float*)(ws + BIG);
  float* B2  = (float*)(ws + 2 * BIG);
  float* B3  = (float*)(ws + 3 * BIG);
  float* OFF = (float*)(ws + 4 * BIG);
  unsigned short* PK = (unsigned short*)(ws + 4 * BIG + OFFB);

  const int PKC  = 4 * 18 * 32 * 16;  // 64->64 conv, u16 elems
  const int PKO  = 2 * 18 * 32 * 16;  // 64->18 conv
  const int PKFO = 2 * 36 * 32 * 16;  // 128->18 conv
  const int PKFD = 4 * 36 * 32 * 16;  // 128->64 deform
  unsigned short* pk_dec = PK;
  unsigned short* pk_off = pk_dec + 4 * (size_t)PKC;
  unsigned short* pk_dcn = pk_off + 4 * (size_t)PKO;
  unsigned short* pk_fo  = pk_dcn + 4 * (size_t)PKC;
  unsigned short* pk_fd  = pk_fo + (size_t)PKFO;

  for (int i = 0; i < 4; ++i) {
    pack_weights<<<(PKC + 255) / 256, 256, 0, stream>>>(
        deconv_w + (size_t)i * 64 * 64 * 9, pk_dec + (size_t)i * PKC, 64, 64, 3, PKC);
    pack_weights<<<(PKO + 255) / 256, 256, 0, stream>>>(
        off_w + (size_t)i * 18 * 64 * 9, pk_off + (size_t)i * PKO, 64, 18, 2, PKO);
    pack_weights<<<(PKC + 255) / 256, 256, 0, stream>>>(
        dcn_w + (size_t)i * 64 * 64 * 9, pk_dcn + (size_t)i * PKC, 64, 64, 2, PKC);
  }
  pack_weights<<<(PKFO + 255) / 256, 256, 0, stream>>>(fin_off_w, pk_fo, 128, 18, 2, PKFO);
  pack_weights<<<(PKFD + 255) / 256, 256, 0, stream>>>(fin_dcn_w, pk_fd, 128, 64, 2, PKFD);

  const int GRID = 1024, BLK = 128;                 // 4096 wave-tiles (M=32)
  const size_t LDS64  = (size_t)4 * 3 * 34 * 64 * 2;   //  52 KB / block
  const size_t LDS128 = (size_t)4 * 3 * 34 * 128 * 2;  // 104 KB / block
  const int NADD = 8 * 64 * 128 * 128;

  // block 0: h=relu(deconv(x_in)) -> B0 ; off -> OFF ; x1=relu(dcn) -> B1
  conv3_wmma<64, 4, true><<<GRID, BLK, LDS64, stream>>>(x_in, nullptr, pk_dec, deconv_b, B0, 64);
  conv3_wmma<64, 2, false><<<GRID, BLK, LDS64, stream>>>(B0, nullptr, pk_off, off_b, OFF, 18);
  dconv_wmma<64, true><<<GRID, BLK, 0, stream>>>(B0, nullptr, OFF, pk_dcn, dcn_b, B1, 64);

  // block 1: in_cur = x1 + 0 = B1 ; x2 -> B2 ; in_prev = B1
  conv3_wmma<64, 4, true><<<GRID, BLK, LDS64, stream>>>(B1, nullptr, pk_dec + PKC, deconv_b + 64, B0, 64);
  conv3_wmma<64, 2, false><<<GRID, BLK, LDS64, stream>>>(B0, nullptr, pk_off + PKO, off_b + 18, OFF, 18);
  dconv_wmma<64, true><<<GRID, BLK, 0, stream>>>(B0, nullptr, OFF, pk_dcn + PKC, dcn_b + 64, B2, 64);

  // block 2: in_cur = B2 + B1 -> B3 ; x3 -> B2 ; in_prev = B3
  addk<<<(NADD + 255) / 256, 256, 0, stream>>>(B2, B1, B3, NADD);
  conv3_wmma<64, 4, true><<<GRID, BLK, LDS64, stream>>>(B3, nullptr, pk_dec + 2 * PKC, deconv_b + 128, B0, 64);
  conv3_wmma<64, 2, false><<<GRID, BLK, LDS64, stream>>>(B0, nullptr, pk_off + 2 * PKO, off_b + 36, OFF, 18);
  dconv_wmma<64, true><<<GRID, BLK, 0, stream>>>(B0, nullptr, OFF, pk_dcn + 2 * PKC, dcn_b + 128, B2, 64);

  // block 3: in_cur = B2 + B3 -> B1 ; x4 -> B3
  addk<<<(NADD + 255) / 256, 256, 0, stream>>>(B2, B3, B1, NADD);
  conv3_wmma<64, 4, true><<<GRID, BLK, LDS64, stream>>>(B1, nullptr, pk_dec + 3 * PKC, deconv_b + 192, B0, 64);
  conv3_wmma<64, 2, false><<<GRID, BLK, LDS64, stream>>>(B0, nullptr, pk_off + 3 * PKO, off_b + 54, OFF, 18);
  dconv_wmma<64, true><<<GRID, BLK, 0, stream>>>(B0, nullptr, OFF, pk_dcn + 3 * PKC, dcn_b + 192, B3, 64);

  // final: x_cat = concat(x4=B3, x_in) handled by two-pointer sources
  conv3_wmma<128, 2, false><<<GRID, BLK, LDS128, stream>>>(B3, x_in, pk_fo, fin_off_b, OFF, 18);
  dconv_wmma<128, false><<<GRID, BLK, 0, stream>>>(B3, x_in, OFF, pk_fd, fin_dcn_b, (float*)d_out, 64);
}
